// Rel_InteractE_55267639165381
// MI455X (gfx1250) — matrix operand order repaired and checked by reading, not compiled, not run on hardware
//
#include <hip/hip_runtime.h>
#include <hip/hip_bf16.h>

// ---------------------------------------------------------------------------
// InteractE forward for MI455X (gfx1250, wave32, WMMA).
// Heavy GEMMs use v_wmma_f32_16x16x32_bf16; all BN stats fp32 & deterministic.
// FC GEMM: K-split 16 ways x 13 N-accumulators -> activations read once.
// Scoring GEMM: 5 N-tiles/wave, nontemporal output stores (write-once stream).
// ---------------------------------------------------------------------------

typedef __attribute__((ext_vector_type(8)))  __bf16 v8bf;
typedef __attribute__((ext_vector_type(16))) __bf16 v16bf;
typedef __attribute__((ext_vector_type(8)))  float  v8f;

#define BN_EPS 1e-5f

#define BSZ    2048
#define DDIM   200
#define OCH    64
#define FLATD  25600      // 64*400
#define NENT   50000
#define KPAD   224        // DDIM padded to 7*32 for WMMA K
#define NPADFC 208        // DDIM padded to 13*16 for WMMA N
#define KCHUNK 16         // K-split factor for FC GEMM (25600/16 = 1600 = 50*32)
#define KSEG   1600

__device__ __forceinline__ unsigned short f2bf(float f) {
    unsigned u = __float_as_uint(f);
    u += 0x7FFFu + ((u >> 16) & 1u);      // round-to-nearest-even
    return (unsigned short)(u >> 16);
}

__device__ __forceinline__ v16bf join16(v8bf lo, v8bf hi) {
    return __builtin_shufflevector(lo, hi, 0,1,2,3,4,5,6,7,8,9,10,11,12,13,14,15);
}

// ---------------------------------------------------------------------------
// K1: bn0 partial sums.  One block per sample; 400 gathered values per block.
// Chequer permutation does not change the multiset, so stats skip the perm.
__global__ void __launch_bounds__(256) k_bn0_partial(
        const float* __restrict__ nf, const float* __restrict__ rf,
        const int* __restrict__ sub, const int* __restrict__ rel,
        float* __restrict__ part) {
    __shared__ float red[256];
    int b = blockIdx.x, tid = threadIdx.x;
    int s = sub[b], r = rel[b];
    float ps = 0.f, pq = 0.f;
    for (int i = tid; i < 2 * DDIM; i += 256) {
        float v = (i < DDIM) ? nf[(size_t)s * DDIM + i]
                             : rf[(size_t)r * DDIM + (i - DDIM)];
        ps += v; pq += v * v;
    }
    red[tid] = ps; __syncthreads();
    for (int st = 128; st > 0; st >>= 1) { if (tid < st) red[tid] += red[tid + st]; __syncthreads(); }
    if (tid == 0) part[b * 2 + 0] = red[0];
    __syncthreads();
    red[tid] = pq; __syncthreads();
    for (int st = 128; st > 0; st >>= 1) { if (tid < st) red[tid] += red[tid + st]; __syncthreads(); }
    if (tid == 0) part[b * 2 + 1] = red[0];
}

// K2: finalize bn0 (deterministic fixed-order reduce of 2048 partials).
__global__ void __launch_bounds__(256) k_bn0_final(
        const float* __restrict__ part, const float* __restrict__ g,
        const float* __restrict__ bb, float* __restrict__ bn0c) {
    __shared__ float red[256];
    int tid = threadIdx.x;
    float ps = 0.f, pq = 0.f;
    for (int b = tid; b < BSZ; b += 256) { ps += part[b * 2]; pq += part[b * 2 + 1]; }
    red[tid] = ps; __syncthreads();
    for (int st = 128; st > 0; st >>= 1) { if (tid < st) red[tid] += red[tid + st]; __syncthreads(); }
    float S = red[0]; __syncthreads();
    red[tid] = pq; __syncthreads();
    for (int st = 128; st > 0; st >>= 1) { if (tid < st) red[tid] += red[tid + st]; __syncthreads(); }
    if (tid == 0) {
        float SQ = red[0];
        const float invN = 1.f / (float)(BSZ * 2 * DDIM);
        float m = S * invN;
        float v = SQ * invN - m * m;
        bn0c[0] = m;
        bn0c[1] = g[0] * rsqrtf(v + BN_EPS);
        bn0c[2] = bb[0];
    }
}

// ---------------------------------------------------------------------------
// K3/K5: per-sample dynamic 3x3 conv (pad 1) over the 20x20 chequer image.
// STATS pass emits per-(block,channel) partial sum/sumsq; WRITE pass applies
// the folded (cbn ∘ bn1) affine + relu and packs bf16 activations.
template <int WRITE>
__global__ void __launch_bounds__(256) k_conv(
        const float* __restrict__ nf, const float* __restrict__ rf,
        const float* __restrict__ fw, const int* __restrict__ sub,
        const int* __restrict__ rel, const float* __restrict__ bn0c,
        float* __restrict__ chpart,            // stats pass
        const float* __restrict__ chScale,     // write pass
        const float* __restrict__ chShift,
        unsigned short* __restrict__ abf) {
    __shared__ float xs[400];
    __shared__ float red[256];
    int b = blockIdx.x, tid = threadIdx.x;
    int s = sub[b], r = rel[b];
    float m0 = bn0c[0], s0 = bn0c[1], b0 = bn0c[2];
    // gather + chequer permute + bn0:  PERM[p] = ((t+r)&1)*200 + t*10 + r/2
    for (int p = tid; p < 400; p += 256) {
        int t = p / 20, rr = p % 20;
        int src = t * 10 + (rr >> 1);
        float v = ((t + rr) & 1) ? rf[(size_t)r * DDIM + src]
                                 : nf[(size_t)s * DDIM + src];
        xs[p] = (v - m0) * s0 + b0;
    }
    __syncthreads();
    const float* fbase = fw + (size_t)r * (OCH * 9);
    for (int oc = 0; oc < OCH; ++oc) {
        float flt[9];
        #pragma unroll
        for (int j = 0; j < 9; ++j) flt[j] = fbase[oc * 9 + j];
        float sc = 0.f, sh = 0.f;
        if (WRITE) { sc = chScale[oc]; sh = chShift[oc]; }
        float ps = 0.f, pq = 0.f;
        for (int p = tid; p < 400; p += 256) {
            int h = p / 20, w = p % 20;
            float acc = 0.f;
            #pragma unroll
            for (int dh = 0; dh < 3; ++dh) {
                int hh = h + dh - 1;
                if ((unsigned)hh < 20u) {
                    #pragma unroll
                    for (int dw = 0; dw < 3; ++dw) {
                        int ww = w + dw - 1;
                        if ((unsigned)ww < 20u) acc += xs[hh * 20 + ww] * flt[dh * 3 + dw];
                    }
                }
            }
            if (WRITE) {
                float yv = fmaxf(sc * acc + sh, 0.f);
                abf[(size_t)b * FLATD + oc * 400 + p] = f2bf(yv);
            } else {
                ps += acc; pq += acc * acc;
            }
        }
        if (!WRITE) {
            red[tid] = ps; __syncthreads();
            for (int st = 128; st > 0; st >>= 1) { if (tid < st) red[tid] += red[tid + st]; __syncthreads(); }
            if (tid == 0) chpart[((size_t)b * OCH + oc) * 2 + 0] = red[0];
            __syncthreads();
            red[tid] = pq; __syncthreads();
            for (int st = 128; st > 0; st >>= 1) { if (tid < st) red[tid] += red[tid + st]; __syncthreads(); }
            if (tid == 0) chpart[((size_t)b * OCH + oc) * 2 + 1] = red[0];
            __syncthreads();
        }
    }
}

// K4: fold cbn + bn1 into one affine per channel (exact composition:
// mean(y1)=cbn_b, var(y1)=g^2 v/(v+eps)).  Fixed-order reduce: deterministic.
__global__ void k_chan_final(
        const float* __restrict__ chpart,
        const float* __restrict__ cg, const float* __restrict__ cb,
        const float* __restrict__ g1, const float* __restrict__ b1,
        float* __restrict__ chScale, float* __restrict__ chShift) {
    int oc = threadIdx.x;
    if (oc >= OCH) return;
    float S = 0.f, SQ = 0.f;
    for (int b = 0; b < BSZ; ++b) {
        S  += chpart[((size_t)b * OCH + oc) * 2 + 0];
        SQ += chpart[((size_t)b * OCH + oc) * 2 + 1];
    }
    const float invN = 1.f / (float)(BSZ * 400);
    float m = S * invN;
    float v = SQ * invN - m * m;
    float inv1 = rsqrtf(v + BN_EPS);
    float gc = cg[oc];
    (void)cb; // cbn_b cancels exactly in the composition
    float var1 = gc * gc * v * inv1 * inv1;
    float inv2 = rsqrtf(var1 + BN_EPS);
    float scale = g1[oc] * gc * inv1 * inv2;
    chScale[oc] = scale;
    chShift[oc] = b1[oc] - scale * m;
}

// ---------------------------------------------------------------------------
// K6: fc_w (200 x 25600 f32) -> bf16, N padded to 208 rows (zeros).
__global__ void __launch_bounds__(256) k_cvt_fcw(const float* __restrict__ fcw,
                                                 unsigned short* __restrict__ dst) {
    size_t idx = (size_t)blockIdx.x * 256 + threadIdx.x;  // NPADFC*FLATD total
    int n = (int)(idx / FLATD);
    size_t k = idx % FLATD;
    float v = (n < DDIM) ? fcw[(size_t)n * FLATD + k] : 0.f;
    dst[idx] = f2bf(v);
}

// K7: FC GEMM partials.  2048 waves: wave = (M-tile tm, K-chunk kc).
// Each wave keeps all 13 N-tile accumulators (104 VGPRs) so the 105 MB
// activation tensor is streamed from HBM exactly once; fc_w^T (bf16, 10.6 MB)
// stays L2-resident.  Partials reduced deterministically in K7b.
__global__ void __launch_bounds__(256) k_fc_gemm(
        const unsigned short* __restrict__ A, const unsigned short* __restrict__ Bm,
        float* __restrict__ Hpart) {
    int wid  = blockIdx.x * 8 + (threadIdx.x >> 5);   // 256 blocks -> 2048 waves
    int tm   = wid & 127;                              // M-tile
    int kc   = wid >> 7;                               // K-chunk 0..15
    int lane = threadIdx.x & 31;
    int l15  = lane & 15;
    int hiHalf = (lane >= 16);
    int kbase = kc * KSEG;
    const unsigned short* aRow = A  + (size_t)(tm * 16 + l15) * FLATD + kbase + (hiHalf ? 8 : 0);
    const unsigned short* bCol = Bm + (size_t)l15 * FLATD + kbase + (hiHalf ? 8 : 0);
    v8f acc[13];
    #pragma unroll
    for (int j = 0; j < 13; ++j)
        acc[j] = (v8f){0.f, 0.f, 0.f, 0.f, 0.f, 0.f, 0.f, 0.f};
    #pragma unroll 1
    for (int ks = 0; ks < KSEG; ks += 32) {
        __builtin_prefetch(aRow + ks + 512, 0, 3);
        v8bf a0 = *(const v8bf*)(aRow + ks);           // A: K = base..base+7
        v8bf a1 = *(const v8bf*)(aRow + ks + 16);      //    K = base+16..base+23
        v16bf av = join16(a0, a1);
        #pragma unroll
        for (int j = 0; j < 13; ++j) {
            const unsigned short* p = bCol + (size_t)j * 16 * FLATD + ks;
            v8bf bl = *(const v8bf*)(p);               // B: K = base..base+7, base+16..base+23
            v8bf bh = *(const v8bf*)(p + 16);
            acc[j] = __builtin_amdgcn_wmma_f32_16x16x32_bf16(
                    false, av, false, join16(bl, bh), (short)0, acc[j], false, false);
        }
    }
    // Hpart[kc][m][n208]
    float* hp = Hpart + (size_t)kc * BSZ * NPADFC;
    #pragma unroll
    for (int j = 0; j < 13; ++j) {
        int n = j * 16 + l15;
        #pragma unroll
        for (int v = 0; v < 8; ++v) {
            int m = tm * 16 + v + (hiHalf ? 8 : 0);
            hp[(size_t)m * NPADFC + n] = acc[j][v];
        }
    }
}

// K7b: deterministic fixed-order reduce of the 16 K-chunk partials + bias.
__global__ void __launch_bounds__(256) k_fc_reduce(
        const float* __restrict__ Hpart, const float* __restrict__ fc_b,
        float* __restrict__ H) {
    size_t idx = (size_t)blockIdx.x * 256 + threadIdx.x;  // BSZ*NPADFC total
    int m = (int)(idx / NPADFC);
    int n = (int)(idx % NPADFC);
    float s = 0.f;
    #pragma unroll
    for (int c = 0; c < KCHUNK; ++c)
        s += Hpart[(size_t)c * BSZ * NPADFC + idx];
    H[(size_t)m * KPAD + n] = (n < DDIM) ? (s + fc_b[n]) : 0.f;
}

// K8: bn2 (1d over batch), one block per column, fixed tree: deterministic.
__global__ void __launch_bounds__(256) k_col_stats(
        const float* __restrict__ H, const float* __restrict__ g2,
        const float* __restrict__ b2, float* __restrict__ colScale,
        float* __restrict__ colShift) {
    __shared__ float red[256];
    int n = blockIdx.x, tid = threadIdx.x;
    float ps = 0.f, pq = 0.f;
    for (int m = tid; m < BSZ; m += 256) {
        float v = H[(size_t)m * KPAD + n];
        ps += v; pq += v * v;
    }
    red[tid] = ps; __syncthreads();
    for (int st = 128; st > 0; st >>= 1) { if (tid < st) red[tid] += red[tid + st]; __syncthreads(); }
    float S = red[0]; __syncthreads();
    red[tid] = pq; __syncthreads();
    for (int st = 128; st > 0; st >>= 1) { if (tid < st) red[tid] += red[tid + st]; __syncthreads(); }
    if (tid == 0) {
        float SQ = red[0];
        float m = S / (float)BSZ;
        float v = SQ / (float)BSZ - m * m;
        float cs = g2[n] * rsqrtf(v + BN_EPS);
        colScale[n] = cs;
        colShift[n] = b2[n] - cs * m;
    }
}

// K9: apply bn2 + relu, pack h -> bf16 with K padded to 224 (zeros).
__global__ void __launch_bounds__(256) k_hb_pack(
        const float* __restrict__ H, const float* __restrict__ colScale,
        const float* __restrict__ colShift, unsigned short* __restrict__ HB) {
    size_t idx = (size_t)blockIdx.x * 256 + threadIdx.x;  // BSZ*KPAD total
    int n = (int)(idx % KPAD);
    if (n < DDIM) {
        float v = fmaxf(colScale[n] * H[idx] + colShift[n], 0.f);
        HB[idx] = f2bf(v);
    } else {
        HB[idx] = 0;
    }
}

// K10: n_feats (50000 x 200 f32) -> bf16, K padded to 224 (zeros).
__global__ void __launch_bounds__(256) k_cvt_nf(const float* __restrict__ nf,
                                                unsigned short* __restrict__ NFB) {
    size_t idx = (size_t)blockIdx.x * 256 + threadIdx.x;  // NENT*KPAD total
    int n = (int)(idx / KPAD);
    int k = (int)(idx % KPAD);
    NFB[idx] = (k < DDIM) ? f2bf(nf[(size_t)n * DDIM + k]) : (unsigned short)0;
}

// K11: scoring GEMM  out[2048 x 50000] = h_bf16 * n_feats_bf16^T + ent_bias.
// 5 N-tiles per wave (3125 = 5*625): A fragment amortized 5x, 35 static WMMAs.
// Output is a write-once 410 MB stream -> nontemporal stores keep NFB/HB in L2.
__global__ void __launch_bounds__(256) k_out_gemm(
        const unsigned short* __restrict__ HB, const unsigned short* __restrict__ NFB,
        const float* __restrict__ ent_bias, float* __restrict__ out) {
    int wave   = threadIdx.x >> 5;
    int mGroup = blockIdx.x & 15;          // 16 groups of 8 M-tiles
    int nGroup = blockIdx.x >> 4;          // 625 groups of 5 N-tiles
    int tm     = mGroup * 8 + wave;
    int lane = threadIdx.x & 31;
    int l15  = lane & 15;
    int hiHalf = (lane >= 16);
    const unsigned short* aRow  = HB  + (size_t)(tm * 16 + l15) * KPAD + (hiHalf ? 8 : 0);
    const unsigned short* bBase = NFB + (size_t)(nGroup * 80 + l15) * KPAD + (hiHalf ? 8 : 0);
    v8f acc[5];
    #pragma unroll
    for (int j = 0; j < 5; ++j)
        acc[j] = (v8f){0.f, 0.f, 0.f, 0.f, 0.f, 0.f, 0.f, 0.f};
    #pragma unroll
    for (int k0 = 0; k0 < KPAD; k0 += 32) {
        v8bf a0 = *(const v8bf*)(aRow + k0);
        v8bf a1 = *(const v8bf*)(aRow + k0 + 16);
        v16bf av = join16(a0, a1);
        #pragma unroll
        for (int j = 0; j < 5; ++j) {
            const unsigned short* p = bBase + (size_t)j * 16 * KPAD + k0;
            v8bf bl = *(const v8bf*)(p);
            v8bf bh = *(const v8bf*)(p + 16);
            acc[j] = __builtin_amdgcn_wmma_f32_16x16x32_bf16(
                    false, av, false, join16(bl, bh), (short)0, acc[j], false, false);
        }
    }
    #pragma unroll
    for (int j = 0; j < 5; ++j) {
        int n = nGroup * 80 + j * 16 + l15;
        float eb = ent_bias[n];
        #pragma unroll
        for (int v = 0; v < 8; ++v) {
            int m = tm * 16 + v + (hiHalf ? 8 : 0);
            __builtin_nontemporal_store(acc[j][v] + eb, &out[(size_t)m * NENT + n]);
        }
    }
}

// ---------------------------------------------------------------------------
extern "C" void kernel_launch(void* const* d_in, const int* in_sizes, int n_in,
                              void* d_out, int out_size, void* d_ws, size_t ws_size,
                              hipStream_t stream) {
    (void)in_sizes; (void)n_in; (void)out_size; (void)ws_size;
    const float* n_feats  = (const float*)d_in[0];
    const float* r_feats  = (const float*)d_in[1];
    const float* filter_w = (const float*)d_in[2];
    const float* bn0_g    = (const float*)d_in[3];
    const float* bn0_b    = (const float*)d_in[4];
    const float* cbn_g    = (const float*)d_in[5];
    const float* cbn_b    = (const float*)d_in[6];
    const float* bn1_g    = (const float*)d_in[7];
    const float* bn1_b    = (const float*)d_in[8];
    const float* fc_w     = (const float*)d_in[9];
    const float* fc_b     = (const float*)d_in[10];
    const float* bn2_g    = (const float*)d_in[11];
    const float* bn2_b    = (const float*)d_in[12];
    const float* ent_bias = (const float*)d_in[13];
    const int*   sub      = (const int*)d_in[14];
    const int*   rel      = (const int*)d_in[15];
    float* out = (float*)d_out;

    char* ws = (char*)d_ws;
    size_t off = 0;
    auto alloc = [&](size_t bytes) {
        size_t r = off;
        off += (bytes + 255) & ~(size_t)255;
        return r;
    };
    float* bn0c     = (float*)(ws + alloc(16 * 4));
    float* part0    = (float*)(ws + alloc((size_t)BSZ * 2 * 4));
    float* chpart   = (float*)(ws + alloc((size_t)BSZ * OCH * 2 * 4));
    float* chScale  = (float*)(ws + alloc(OCH * 4));
    float* chShift  = (float*)(ws + alloc(OCH * 4));
    float* colScale = (float*)(ws + alloc(KPAD * 4));
    float* colShift = (float*)(ws + alloc(KPAD * 4));
    float* H        = (float*)(ws + alloc((size_t)BSZ * KPAD * 4));
    float* Hpart    = (float*)(ws + alloc((size_t)KCHUNK * BSZ * NPADFC * 4));
    unsigned short* ABF = (unsigned short*)(ws + alloc((size_t)BSZ * FLATD * 2));
    unsigned short* FWB = (unsigned short*)(ws + alloc((size_t)NPADFC * FLATD * 2));
    unsigned short* HB  = (unsigned short*)(ws + alloc((size_t)BSZ * KPAD * 2));
    unsigned short* NFB = (unsigned short*)(ws + alloc((size_t)NENT * KPAD * 2));

    k_bn0_partial<<<BSZ, 256, 0, stream>>>(n_feats, r_feats, sub, rel, part0);
    k_bn0_final<<<1, 256, 0, stream>>>(part0, bn0_g, bn0_b, bn0c);
    k_conv<0><<<BSZ, 256, 0, stream>>>(n_feats, r_feats, filter_w, sub, rel, bn0c,
                                       chpart, nullptr, nullptr, nullptr);
    k_chan_final<<<1, 64, 0, stream>>>(chpart, cbn_g, cbn_b, bn1_g, bn1_b,
                                       chScale, chShift);
    k_conv<1><<<BSZ, 256, 0, stream>>>(n_feats, r_feats, filter_w, sub, rel, bn0c,
                                       nullptr, chScale, chShift, ABF);
    k_cvt_fcw<<<(NPADFC * FLATD) / 256, 256, 0, stream>>>(fc_w, FWB);
    k_fc_gemm<<<256, 256, 0, stream>>>(ABF, FWB, Hpart);               // 2048 waves
    k_fc_reduce<<<(BSZ * NPADFC) / 256, 256, 0, stream>>>(Hpart, fc_b, H);
    k_col_stats<<<DDIM, 256, 0, stream>>>(H, bn2_g, bn2_b, colScale, colShift);
    k_hb_pack<<<(BSZ * KPAD) / 256, 256, 0, stream>>>(H, colScale, colShift, HB);
    k_cvt_nf<<<(NENT * KPAD) / 256, 256, 0, stream>>>(n_feats, NFB);
    k_out_gemm<<<16 * 625, 256, 0, stream>>>(HB, NFB, ent_bias, out);  // 80000 waves
}
